// SupCon_hcl_49323404427556
// MI455X (gfx1250) — matrix operand order repaired and checked by reading, not compiled, not run on hardware
//
#include <hip/hip_runtime.h>
#include <hip/hip_bf16.h>
#include <math.h>

typedef __attribute__((ext_vector_type(16))) __bf16 v16bf;
typedef __attribute__((ext_vector_type(8)))  __bf16 v8bf;
typedef __attribute__((ext_vector_type(8)))  float  v8f;

#define B_ROWS 4096
#define NROWS  8192
#define D      512

__device__ __forceinline__ float wave_red_sum32(float v) {
#pragma unroll
  for (int off = 1; off < 32; off <<= 1) v += __shfl_xor(v, off, 32);
  return v;
}

__device__ __forceinline__ v16bf cat16(v8bf a, v8bf b) {
  return __builtin_shufflevector(a, b, 0, 1, 2, 3, 4, 5, 6, 7, 8, 9, 10, 11, 12, 13, 14, 15);
}

#define WMMA_BF16(A, Bm, C) \
  __builtin_amdgcn_wmma_f32_16x16x32_bf16(false, (A), false, (Bm), (short)0, (C), false, false)

// ---- kernel 1: L2-normalize rows of [out_1;out_2], split into bf16 hi/lo, zero row accumulators
__global__ void __launch_bounds__(128) k_norm(const float* __restrict__ o1,
                                              const float* __restrict__ o2,
                                              __bf16* __restrict__ hi,
                                              __bf16* __restrict__ lo,
                                              float* __restrict__ sumA,
                                              float* __restrict__ sumB) {
  const int row = blockIdx.x;
  const float* src = (row < B_ROWS) ? (o1 + (size_t)row * D)
                                    : (o2 + (size_t)(row - B_ROWS) * D);
  const int tid = threadIdx.x;               // 128 threads * 4 floats = 512
  float4 x = ((const float4*)src)[tid];
  float ss = x.x * x.x + x.y * x.y + x.z * x.z + x.w * x.w;
  ss = wave_red_sum32(ss);
  __shared__ float red[4];
  const int wid = tid >> 5, lane = tid & 31;
  if (lane == 0) red[wid] = ss;
  __syncthreads();
  const float invn = 1.0f / sqrtf(red[0] + red[1] + red[2] + red[3]);
  __bf16* ho = hi + (size_t)row * D + tid * 4;
  __bf16* lw = lo + (size_t)row * D + tid * 4;
  float vs[4] = {x.x, x.y, x.z, x.w};
#pragma unroll
  for (int i = 0; i < 4; ++i) {
    float v = vs[i] * invn;
    __bf16 h = (__bf16)v;                    // RNE
    ho[i] = h;
    lw[i] = (__bf16)(v - (float)h);          // residual
  }
  if (tid == 0) { sumA[row] = 0.0f; sumB[row] = 0.0f; }
}

// ---- kernel 2: t_pos(i) = 2*dot(o1n_i, o2n_i); one wave per i
__global__ void __launch_bounds__(256) k_tpos(const __bf16* __restrict__ hi,
                                              const __bf16* __restrict__ lo,
                                              float* __restrict__ tpos) {
  const int wid = threadIdx.x >> 5, lane = threadIdx.x & 31;
  const int i = blockIdx.x * 8 + wid;
  const __bf16* a_h = hi + (size_t)i * D;
  const __bf16* a_l = lo + (size_t)i * D;
  const __bf16* b_h = hi + (size_t)(i + B_ROWS) * D;
  const __bf16* b_l = lo + (size_t)(i + B_ROWS) * D;
  float s = 0.0f;
  for (int k = lane; k < D; k += 32) {
    float a = (float)a_h[k] + (float)a_l[k];
    float b = (float)b_h[k] + (float)b_l[k];
    s = fmaf(a, b, s);
  }
  s = wave_red_sum32(s);
  if (lane == 0) tpos[i] = 2.0f * s;
}

// ---- kernel 3: fused Gram-matrix + exp + row-sum via bf16-split WMMA
// wave = 32-row block (two 16-row stripes sharing B tiles), 4 col-tiles per iteration:
// per k-step 8 A-loads + 16 B-loads feed 24 WMMAs (1:1 vmem:wmma issue).
__global__ void __launch_bounds__(128) k_gemm(const __bf16* __restrict__ hi,
                                              const __bf16* __restrict__ lo,
                                              float* __restrict__ sumA,
                                              float* __restrict__ sumB) {
  const int lane = threadIdx.x & 31;
  const int wid  = threadIdx.x >> 5;
  const int half = lane >> 4;                // 0: lanes 0-15, 1: lanes 16-31
  const int lr   = lane & 15;
  const int m0   = (blockIdx.x * 4 + wid) * 32;      // 32 rows per wave
  const int cbeg = blockIdx.y * (NROWS / 8);         // 8-way column split
  const int cend = cbeg + (NROWS / 8);

  // A 16x32 bf16 layout: lane holds row m+lr; lanes 0-15 K chunks {0..7,16..23}, lanes 16-31 {8..15,24..31}
  const __bf16* Ah0 = hi + (size_t)(m0 + lr) * D;
  const __bf16* Al0 = lo + (size_t)(m0 + lr) * D;
  const __bf16* Ah1 = hi + (size_t)(m0 + 16 + lr) * D;
  const __bf16* Al1 = lo + (size_t)(m0 + 16 + lr) * D;
  const int aoff = half * 8;

  float rsA[2][8], rsB[2][8];
#pragma unroll
  for (int s = 0; s < 2; ++s)
#pragma unroll
    for (int v = 0; v < 8; ++v) { rsA[s][v] = 0.0f; rsB[s][v] = 0.0f; }

  for (int n0 = cbeg; n0 < cend; n0 += 64) {
    v8f acc0[4] = {};
    v8f acc1[4] = {};
    for (int k0 = 0; k0 < D; k0 += 32) {
      v16bf Ah_0 = cat16(*(const v8bf*)(Ah0 + k0 + aoff),
                         *(const v8bf*)(Ah0 + k0 + 16 + aoff));
      v16bf Al_0 = cat16(*(const v8bf*)(Al0 + k0 + aoff),
                         *(const v8bf*)(Al0 + k0 + 16 + aoff));
      v16bf Ah_1 = cat16(*(const v8bf*)(Ah1 + k0 + aoff),
                         *(const v8bf*)(Ah1 + k0 + 16 + aoff));
      v16bf Al_1 = cat16(*(const v8bf*)(Al1 + k0 + aoff),
                         *(const v8bf*)(Al1 + k0 + 16 + aoff));
#pragma unroll
      for (int t = 0; t < 4; ++t) {
        // B 32x16 layout: lane holds column n0+16t+lr (symmetric Gram => a row of Out),
        // K chunks {half*8..+7, 16+half*8..+7} (same as A)
        const __bf16* bp_h = hi + (size_t)(n0 + t * 16 + lr) * D + k0;
        const __bf16* bp_l = lo + (size_t)(n0 + t * 16 + lr) * D + k0;
        v16bf Bh = cat16(*(const v8bf*)(bp_h + half * 8), *(const v8bf*)(bp_h + 16 + half * 8));
        v16bf Bl = cat16(*(const v8bf*)(bp_l + half * 8), *(const v8bf*)(bp_l + 16 + half * 8));
        // fp32-via-bf16 split: hi*hi + hi*lo + lo*hi, for both row stripes
        acc0[t] = WMMA_BF16(Ah_0, Bh, acc0[t]);
        acc0[t] = WMMA_BF16(Ah_0, Bl, acc0[t]);
        acc0[t] = WMMA_BF16(Al_0, Bh, acc0[t]);
        acc1[t] = WMMA_BF16(Ah_1, Bh, acc1[t]);
        acc1[t] = WMMA_BF16(Ah_1, Bl, acc1[t]);
        acc1[t] = WMMA_BF16(Al_1, Bh, acc1[t]);
      }
    }
    // acc[t][v] = dot(row, col); t_ij = 2*dot, so exp(1.5t)=exp(3*dot), exp(0.5t)=exp(dot)
#pragma unroll
    for (int t = 0; t < 4; ++t)
#pragma unroll
      for (int v = 0; v < 8; ++v) {
        float e0 = acc0[t][v];
        float e1 = acc1[t][v];
        rsA[0][v] += __expf(3.0f * e0);
        rsB[0][v] += __expf(e0);
        rsA[1][v] += __expf(3.0f * e1);
        rsB[1][v] += __expf(e1);
      }
  }

  // C layout: VGPR v, lanes 0-15 -> row base+v; lanes 16-31 -> row base+8+v. Half-wave reduce.
#pragma unroll
  for (int s = 0; s < 2; ++s)
#pragma unroll
    for (int v = 0; v < 8; ++v) {
      float a = rsA[s][v], b = rsB[s][v];
#pragma unroll
      for (int off = 1; off < 16; off <<= 1) {
        a += __shfl_xor(a, off, 32);
        b += __shfl_xor(b, off, 32);
      }
      if (lr == 0) {
        const int row = m0 + s * 16 + half * 8 + v;
        atomicAdd(&sumA[row], a);
        atomicAdd(&sumB[row], b);
      }
    }
}

// ---- kernel 4: subtract self/positive terms analytically, assemble NCE loss
__global__ void __launch_bounds__(256) k_final(const float* __restrict__ sumA,
                                               const float* __restrict__ sumB,
                                               const float* __restrict__ tpos,
                                               float* __restrict__ out) {
  __shared__ float red[256];
  float part = 0.0f;
  for (int r = threadIdx.x; r < NROWS; r += 256) {
    const float tp = tpos[r & (B_ROWS - 1)];
    const float SA = sumA[r] - expf(3.0f) - expf(1.5f * tp);  // drop j=i (dot=1) and j=pos
    const float SB = sumB[r] - expf(1.0f) - expf(0.5f * tp);
    const float negsum = SA / SB;
    const float denom = expf(tp) + (float)(NROWS - 2) * negsum;
    part += tp - logf(denom);                                  // log(pos/denom)
  }
  red[threadIdx.x] = part;
  __syncthreads();
  for (int s = 128; s > 0; s >>= 1) {
    if (threadIdx.x < s) red[threadIdx.x] += red[threadIdx.x + s];
    __syncthreads();
  }
  if (threadIdx.x == 0) out[0] = -red[0] / (float)NROWS;
}

extern "C" void kernel_launch(void* const* d_in, const int* in_sizes, int n_in,
                              void* d_out, int out_size, void* d_ws, size_t ws_size,
                              hipStream_t stream) {
  (void)in_sizes; (void)n_in; (void)out_size; (void)ws_size;
  // d_in[0]=features (unused), d_in[1]=out_1, d_in[2]=out_2, d_in[3]=indexes (arange, unused)
  const float* out1 = (const float*)d_in[1];
  const float* out2 = (const float*)d_in[2];

  char* ws = (char*)d_ws;
  __bf16* hi  = (__bf16*)ws;                                     // 8 MB
  __bf16* lo  = (__bf16*)(ws + (size_t)NROWS * D * 2);           // 8 MB
  float* sumA = (float*)(ws + (size_t)NROWS * D * 4);            // 32 KB
  float* sumB = sumA + NROWS;                                    // 32 KB
  float* tpos = sumB + NROWS;                                    // 16 KB
  float* out  = (float*)d_out;

  k_norm<<<NROWS, 128, 0, stream>>>(out1, out2, hi, lo, sumA, sumB);
  k_tpos<<<B_ROWS / 8, 256, 0, stream>>>(hi, lo, tpos);
  k_gemm<<<dim3(NROWS / 128, 8), 128, 0, stream>>>(hi, lo, sumA, sumB);
  k_final<<<1, 256, 0, stream>>>(sumA, sumB, tpos, out);
}
